// FM_Layer_81664508166721
// MI455X (gfx1250) — matrix operand order repaired and checked by reading, not compiled, not run on hardware
//
#include <hip/hip_runtime.h>
#include <hip/hip_bf16.h>

typedef __bf16 bf16_t;
typedef __attribute__((ext_vector_type(16))) __bf16 v16bf;
typedef __attribute__((ext_vector_type(8)))  float  v8f;

#define D_DIM        512
#define ROWS_PER_WG  128
#define TOTAL_ROWS   32768                     // B*N = 256*128
#define NUM_WG       (TOTAL_ROWS / ROWS_PER_WG) // 256
#define BS_STRIDE    40                        // ushorts per e-row: 32 data + 8 pad = 80B (16B aligned)

// workspace layout (floats):
//   wsf[0]          interaction scalar
//   wsf[16..16+256) per-WG partials
//   wsf[512..1024)  s[d] = sum_e V[d][e]^2

// ---------------- kernel 0: s[d] = sum_e V[d][e]^2 ----------------
__global__ __launch_bounds__(256) void fm_prep(const float* __restrict__ V,
                                               float* __restrict__ s) {
    const int lane = threadIdx.x & 31;
    const int wave = threadIdx.x >> 5;
    const int d    = blockIdx.x * 8 + wave;          // 64 blocks * 8 waves = 512 rows
    const float* rp = V + (size_t)d * D_DIM;
    float a = 0.f;
#pragma unroll
    for (int j = 0; j < 16; ++j) { float v = rp[lane + 32 * j]; a += v * v; }
#pragma unroll
    for (int off = 16; off >= 1; off >>= 1) a += __shfl_xor(a, off, 32);
    if (lane == 0) s[d] = a;
}

// A fragment (contiguous k order used by this code, same as B): lane half h reads row bytes
// K in [kbase, kbase+16), kbase = ks*32 + 16*h.
__device__ __forceinline__ v16bf load_a_frag(const float* __restrict__ rowp, int kbase) {
    float4 f0 = *(const float4*)(rowp + kbase);
    float4 f1 = *(const float4*)(rowp + kbase + 4);
    float4 f2 = *(const float4*)(rowp + kbase + 8);
    float4 f3 = *(const float4*)(rowp + kbase + 12);
    v16bf a;
    a[0]=(bf16_t)f0.x;  a[1]=(bf16_t)f0.y;  a[2]=(bf16_t)f0.z;  a[3]=(bf16_t)f0.w;
    a[4]=(bf16_t)f1.x;  a[5]=(bf16_t)f1.y;  a[6]=(bf16_t)f1.z;  a[7]=(bf16_t)f1.w;
    a[8]=(bf16_t)f2.x;  a[9]=(bf16_t)f2.y;  a[10]=(bf16_t)f2.z; a[11]=(bf16_t)f2.w;
    a[12]=(bf16_t)f3.x; a[13]=(bf16_t)f3.y; a[14]=(bf16_t)f3.z; a[15]=(bf16_t)f3.w;
    return a;
}

// B fragment (contiguous k order used by this code, same as A): lane = column N, lane half h holds the
// contiguous K block [16h, 16h+16) -> 32 contiguous bytes in our [e][k] panel.
__device__ __forceinline__ v16bf load_b_frag(const unsigned short* p) {
    union { uint4 u[2]; v16bf v; } U;
    U.u[0] = *(const uint4*)(p);
    U.u[1] = *(const uint4*)(p + 8);
    return U.v;
}

// ---------------- kernel 1: linear part + 0.5*x^2 s  and  sum ||x V||^2 ----------------
__global__ __launch_bounds__(256) void fm_main(const float* __restrict__ x,
                                               const float* __restrict__ W,
                                               const float* __restrict__ bptr,
                                               const float* __restrict__ V,
                                               float* __restrict__ out,
                                               const float* __restrict__ s,
                                               float* __restrict__ partials) {
    __shared__ unsigned short Bs[128 * BS_STRIDE];   // 10,240 B, padded for b128 reads
    __shared__ float wred[8];

    const int tid   = threadIdx.x;
    const int wave  = tid >> 5;
    const int lane  = tid & 31;
    const int lrow  = lane & 15;      // A row / B column within tile
    const int khalf = lane >> 4;
    const int row0  = blockIdx.x * ROWS_PER_WG;

    // ---- pass 1: linear part -> out, and per-lane sum of x^2 * s[d]
    float acc_s = 0.f;
    const float b0 = bptr[0];
    for (int i = 0; i < 16; ++i) {
        const int r = wave * 16 + i;
        const float* rp = x + (size_t)(row0 + r) * D_DIM;
        float acc_l = 0.f;
#pragma unroll
        for (int j = 0; j < 16; ++j) {
            const int c = lane + 32 * j;
            const float v = rp[c];
            acc_l += v * W[c];
            acc_s += v * v * s[c];
        }
#pragma unroll
        for (int off = 16; off >= 1; off >>= 1) acc_l += __shfl_xor(acc_l, off, 32);
        if (lane == 0) out[row0 + r] = acc_l + b0;
    }

    // ---- pass 2: y2 = sum of squares of (X_strip * V) via bf16 WMMA
    float y2 = 0.f;
    const float* arowp = x + (size_t)(row0 + wave * 16 + lrow) * D_DIM;
    const v8f vzero = {0.f, 0.f, 0.f, 0.f, 0.f, 0.f, 0.f, 0.f};

    for (int eb = 0; eb < 4; ++eb) {               // 4 blocks of 128 output columns
        v8f acc[8];
#pragma unroll
        for (int t = 0; t < 8; ++t) acc[t] = vzero;
        const int e0 = eb * 128;

        for (int ks = 0; ks < 16; ++ks) {          // K = 512 in steps of 32
            __syncthreads();                       // previous panel fully consumed
            // stage V[ks*32 .. +32)[e0 .. e0+128) transposed -> Bs[e][k] as bf16
            const float* vbase = V + (size_t)(ks * 32) * D_DIM + e0;
#pragma unroll
            for (int it = 0; it < 16; ++it) {
                const int i  = tid + it * 256;     // 4096 elements total
                const int kk = i >> 7;
                const int e  = i & 127;
                Bs[e * BS_STRIDE + kk] =
                    __builtin_bit_cast(unsigned short, (bf16_t)vbase[(size_t)kk * D_DIM + e]);
            }
            __syncthreads();

            const v16bf a = load_a_frag(arowp, ks * 32 + khalf * 16);
#pragma unroll
            for (int t = 0; t < 8; ++t) {
                const v16bf bf = load_b_frag(Bs + (t * 16 + lrow) * BS_STRIDE + khalf * 16);
                acc[t] = __builtin_amdgcn_wmma_f32_16x16x32_bf16(
                    false, a, false, bf, (short)0, acc[t], false, false);
            }
        }
#pragma unroll
        for (int t = 0; t < 8; ++t)
#pragma unroll
            for (int j = 0; j < 8; ++j) y2 += acc[t][j] * acc[t][j];
    }

    // ---- per-WG partial of the interaction: 0.5*sum(x^2 s) - sum||xV||^2
    float p = 0.5f * acc_s - y2;
#pragma unroll
    for (int off = 16; off >= 1; off >>= 1) p += __shfl_xor(p, off, 32);
    if (lane == 0) wred[wave] = p;
    __syncthreads();
    if (tid == 0) {
        float t = 0.f;
#pragma unroll
        for (int w = 0; w < 8; ++w) t += wred[w];
        partials[blockIdx.x] = t;
    }
}

// ---------------- kernel 2a: deterministic tree-reduce of 256 partials ----------------
__global__ __launch_bounds__(256) void fm_reduce(const float* __restrict__ partials,
                                                 float* __restrict__ inter) {
    __shared__ float sm[256];
    const int tid = threadIdx.x;
    sm[tid] = partials[tid];
    __syncthreads();
    for (int off = 128; off >= 1; off >>= 1) {
        if (tid < off) sm[tid] += sm[tid + off];
        __syncthreads();
    }
    if (tid == 0) inter[0] = sm[0];
}

// ---------------- kernel 2b: out[i] += interaction ----------------
__global__ __launch_bounds__(256) void fm_finish(float* __restrict__ out,
                                                 const float* __restrict__ inter,
                                                 int n) {
    const int i = blockIdx.x * 256 + threadIdx.x;
    if (i < n) out[i] += inter[0];
}

extern "C" void kernel_launch(void* const* d_in, const int* in_sizes, int n_in,
                              void* d_out, int out_size, void* d_ws, size_t ws_size,
                              hipStream_t stream) {
    (void)in_sizes; (void)n_in; (void)ws_size;
    const float* x = (const float*)d_in[0];   // [256,128,512]
    const float* W = (const float*)d_in[1];   // [1,512]
    const float* b = (const float*)d_in[2];   // [1]
    const float* V = (const float*)d_in[3];   // [512,512]
    float* out = (float*)d_out;               // [32768,1]

    float* wsf      = (float*)d_ws;
    float* inter    = wsf;          // 1 float
    float* partials = wsf + 16;     // 256 floats
    float* s        = wsf + 512;    // 512 floats

    fm_prep  <<<64,      256, 0, stream>>>(V, s);
    fm_main  <<<NUM_WG,  256, 0, stream>>>(x, W, b, V, out, s, partials);
    fm_reduce<<<1,       256, 0, stream>>>(partials, inter);
    fm_finish<<<(out_size + 255) / 256, 256, 0, stream>>>(out, inter, out_size);
}
